// LSTM101_20830591386029
// MI455X (gfx1250) — matrix operand order repaired and checked by reading, not compiled, not run on hardware
//
#include <hip/hip_runtime.h>
#include <hip/hip_bf16.h>

// ---------------------------------------------------------------------------
// Fused 2-layer LSTM + projection for MI455X (gfx1250, wave32, WMMA).
//
//   Layer1: H1=60 (pad 64)  -> gates padded to 256 cols = 16 N-tiles of 16
//   Layer2: H2=30 (pad 32)  -> gates padded to 128 cols =  8 N-tiles of 16
//
// Grid: 32 blocks (one per 16-row batch tile), 128 threads (4 waves).
// Wave w owns hidden slice [16w,16w+16): its 4 gate tiles for layer 1
// (tiles 4g+w, g=0..3) and, for w<2, layer 2 (tiles 2g+w).
// Weights are packed once per launch into per-lane WMMA B-fragment layout
// in d_ws and kept in VGPRs across all 2048 timesteps.
// ---------------------------------------------------------------------------

typedef __attribute__((ext_vector_type(16))) _Float16 v16h;
typedef __attribute__((ext_vector_type(8)))  _Float16 v8h;
typedef __attribute__((ext_vector_type(8)))  float    v8f;

#define TSEQ   2048
#define NBATCH 512

// ---------------------------------------------------------------------------
// Weight packing: f32 -> f16 B-fragments (D = A x B, B is K x N tile 32x16).
// B-fragment per-lane layout (16-bit): lane l holds column n = l&15,
// halves h=0..7 map to K = (l>>4)*8 + h, h=8..15 to K = 16 + (l>>4)*8 + (h-8).
// Fragment storage: frags[fragIdx*512 + lane*16 + h].
//   frags[ 0..16)  : W1x  (Wih1^T), 1 K-chunk, 16 N-tiles      (K real 15)
//   frags[16..48)  : W1h  (Whh1^T), 2 K-chunks x 16 N-tiles    (K real 60)
//   frags[48..64)  : W2x  (Wih2^T), 2 K-chunks x  8 N-tiles    (K real 60)
//   frags[64..72)  : W2h  (Whh2^T), 1 K-chunk,  8 N-tiles      (K real 30)
// biases (f32): [0..256) b1, [256..384) b2, [384..416) Wp-padded, [416] bp
// Padded gate col n: layer1 gate=n>>6 unit=n&63 (valid u<60);
//                    layer2 gate=n>>5 unit=n&31 (valid u<30).
// ---------------------------------------------------------------------------
__global__ __launch_bounds__(256) void lstm_pack_kernel(
    const float* __restrict__ Wih1, const float* __restrict__ Whh1,
    const float* __restrict__ bih1, const float* __restrict__ bhh1,
    const float* __restrict__ Wih2, const float* __restrict__ Whh2,
    const float* __restrict__ bih2, const float* __restrict__ bhh2,
    const float* __restrict__ Wp,   const float* __restrict__ bp,
    _Float16* __restrict__ frags,   float* __restrict__ biases)
{
  const int tid = blockIdx.x * 256 + threadIdx.x;
  const int NFRAG_ELEMS = 72 * 512;
  if (tid < NFRAG_ELEMS) {
    const int frag   = tid >> 9;
    const int idx    = tid & 511;
    const int lane   = idx >> 4;
    const int h      = idx & 15;
    const int klocal = ((h >> 3) << 4) + ((lane >> 4) << 3) + (h & 7);   // K position within 32-chunk
    const int nlocal = lane & 15;
    float val = 0.0f;
    if (frag < 16) {                       // W1x
      const int nt = frag;
      const int n = nt * 16 + nlocal, g = n >> 6, u = n & 63, k = klocal;
      if (u < 60 && k < 15) val = Wih1[(g * 60 + u) * 15 + k];
    } else if (frag < 48) {                // W1h
      const int f = frag - 16, nt = f >> 1, kc = f & 1;
      const int n = nt * 16 + nlocal, g = n >> 6, u = n & 63, k = kc * 32 + klocal;
      if (u < 60 && k < 60) val = Whh1[(g * 60 + u) * 60 + k];
    } else if (frag < 64) {                // W2x
      const int f = frag - 48, nt = f >> 1, kc = f & 1;
      const int n = nt * 16 + nlocal, g = n >> 5, u = n & 31, k = kc * 32 + klocal;
      if (u < 30 && k < 60) val = Wih2[(g * 30 + u) * 60 + k];
    } else {                               // W2h
      const int nt = frag - 64;
      const int n = nt * 16 + nlocal, g = n >> 5, u = n & 31, k = klocal;
      if (u < 30 && k < 30) val = Whh2[(g * 30 + u) * 30 + k];
    }
    frags[(size_t)frag * 512 + lane * 16 + h] = (_Float16)val;
  } else if (tid < NFRAG_ELEMS + 512) {
    const int j = tid - NFRAG_ELEMS;
    if (j < 256) {
      const int g = j >> 6, u = j & 63;
      biases[j] = (u < 60) ? (bih1[g * 60 + u] + bhh1[g * 60 + u]) : 0.0f;
    } else if (j < 384) {
      const int n = j - 256, g = n >> 5, u = n & 31;
      biases[j] = (u < 30) ? (bih2[g * 30 + u] + bhh2[g * 30 + u]) : 0.0f;
    } else if (j < 416) {
      const int u = j - 384;
      biases[j] = (u < 30) ? Wp[u] : 0.0f;
    } else if (j == 416) {
      biases[416] = bp[0];
    }
  }
}

// ---------------------------------------------------------------------------
// Device helpers
// ---------------------------------------------------------------------------
__device__ __forceinline__ v8f wmma16(v16h a, v16h b, v8f c) {
  return __builtin_amdgcn_wmma_f32_16x16x32_f16(
      /*neg_a=*/false, a, /*neg_b=*/false, b,
      /*c_mod=*/(short)0, c, /*reuse_a=*/false, /*reuse_b=*/false);
}

__device__ __forceinline__ float sigm_f(float x) {
  return 1.0f / (1.0f + __expf(-x));
}
__device__ __forceinline__ float tanh_f(float x) {
  x = fminf(fmaxf(x, -15.0f), 15.0f);
  const float e = __expf(2.0f * x);
  return (e - 1.0f) / (e + 1.0f);
}

// Build a 16-bit A-fragment (16xK tile, K-chunk base kc in halves) from a
// row-major LDS buffer [16][pitch] halves. Per ISA layout: lanes 0-15 carry
// K = {kc+0..7, kc+16..23}, lanes 16-31 carry K = {kc+8..15, kc+24..31}.
__device__ __forceinline__ v16h lds_a_frag(const _Float16* base, int row,
                                           int pitch, int kc, int g2) {
  const _Float16* p0 = base + row * pitch + kc + g2 * 8;
  v8h c0 = *(const v8h*)p0;          // 16B-aligned
  v8h c1 = *(const v8h*)(p0 + 16);   // 16B-aligned
  v16h a;
#pragma unroll
  for (int i = 0; i < 8; i++) { a[i] = c0[i]; a[i + 8] = c1[i]; }
  return a;
}

// ---------------------------------------------------------------------------
// Fused LSTM kernel
// ---------------------------------------------------------------------------
__global__ __launch_bounds__(128) void lstm_fused_kernel(
    const float* __restrict__ x,          // [512, 2048, 15]
    const _Float16* __restrict__ frags,   // packed B-fragments
    const float* __restrict__ biases,     // packed biases / Wp / bp
    float* __restrict__ out)              // [512, 2048, 1]
{
  __shared__ __align__(32) _Float16 h1s[16 * 64];  // h1 tile, f16, row-major
  __shared__ __align__(32) _Float16 h2s[16 * 32];  // h2 tile, f16, row-major
  __shared__ __align__(16) float    h2f[32 * 16];  // h2 tile, f32, [unit][b]

  const int tidb   = threadIdx.x;
  const int w      = tidb >> 5;        // wave id 0..3
  const int lane   = tidb & 31;
  const int g2     = lane >> 4;        // lane group
  const int nl     = lane & 15;
  const int batch0 = blockIdx.x * 16;
  const int w2id   = w & 1;

  // ---- persistent weight fragments (stay in VGPRs for all 2048 steps) ----
  v16h w1x[4], w1h[4][2], w2x[4][2], w2h[4];
  {
    const _Float16* fb = frags + lane * 16;
#pragma unroll
    for (int g = 0; g < 4; g++) {
      w1x[g]    = *(const v16h*)(fb + (size_t)(4 * g + w) * 512);
      w1h[g][0] = *(const v16h*)(fb + (size_t)(16 + (4 * g + w) * 2 + 0) * 512);
      w1h[g][1] = *(const v16h*)(fb + (size_t)(16 + (4 * g + w) * 2 + 1) * 512);
      w2x[g][0] = *(const v16h*)(fb + (size_t)(48 + (2 * g + w2id) * 2 + 0) * 512);
      w2x[g][1] = *(const v16h*)(fb + (size_t)(48 + (2 * g + w2id) * 2 + 1) * 512);
      w2h[g]    = *(const v16h*)(fb + (size_t)(64 + 2 * g + w2id) * 512);
    }
  }
  float b1v[4], b2v[4];
#pragma unroll
  for (int g = 0; g < 4; g++) {
    b1v[g] = biases[64 * g + 16 * w + nl];
    b2v[g] = biases[256 + 32 * g + 16 * w2id + nl];
  }
  float wpv[30];
#pragma unroll
  for (int u = 0; u < 30; u++) wpv[u] = biases[384 + u];
  const float bpv = biases[416];

  // zero initial state
  for (int i = tidb; i < 16 * 64; i += 128) h1s[i] = (_Float16)0.0f;
  for (int i = tidb; i < 16 * 32; i += 128) h2s[i] = (_Float16)0.0f;
  for (int i = tidb; i < 32 * 16; i += 128) h2f[i] = 0.0f;
  v8f c1 = {}, c2 = {};
  __syncthreads();

  const float* xrow = x + ((size_t)(batch0 + nl) * TSEQ) * 15;
  const int kbase = g2 * 8;

  for (int t = 0; t < TSEQ; t++) {
    // -------- phase A: read x[t] and previous h1/h2 fragments --------
    v16h ax;
#pragma unroll
    for (int i = 0; i < 8; i++) {
      const int K = kbase + i;
      float v = xrow[(K < 15) ? K : 0];
      if (K >= 15) v = 0.0f;
      ax[i]     = (_Float16)v;
      ax[i + 8] = (_Float16)0.0f;   // K 16..31 are padding
    }
    __builtin_prefetch(xrow + 15, 0, 0);         // next timestep row
    v16h a1k0 = lds_a_frag(h1s, nl, 64,  0, g2);
    v16h a1k1 = lds_a_frag(h1s, nl, 64, 32, g2);
    v16h a2   = lds_a_frag(h2s, nl, 32,  0, g2);
    __syncthreads();   // B1: everyone done reading old h1/h2

    // -------- phase B: layer-1 gates (WMMA) + LSTM cell update --------
    v8f acc[4];
#pragma unroll
    for (int g = 0; g < 4; g++) {
      v8f a;
#pragma unroll
      for (int r = 0; r < 8; r++) a[r] = b1v[g];
      a = wmma16(ax,   w1x[g],    a);
      a = wmma16(a1k0, w1h[g][0], a);
      a = wmma16(a1k1, w1h[g][1], a);
      acc[g] = a;
    }
#pragma unroll
    for (int r = 0; r < 8; r++) {
      const float iv = sigm_f(acc[0][r]);
      const float fv = sigm_f(acc[1][r]);
      const float gv = tanh_f(acc[2][r]);
      const float ov = sigm_f(acc[3][r]);
      const float c  = fv * c1[r] + iv * gv;
      c1[r] = c;
      const float hh = ov * tanh_f(c);
      const int b = r + 8 * g2;                   // C/D layout: M = r + 8*(lane/16)
      h1s[b * 64 + 16 * w + nl] = (_Float16)hh;
    }
    __syncthreads();   // B2: new h1 visible

    // -------- phase C: layer-2 gates (waves 0,1) --------
    if (w < 2) {
      v16h n1k0 = lds_a_frag(h1s, nl, 64,  0, g2);
      v16h n1k1 = lds_a_frag(h1s, nl, 64, 32, g2);
      v8f acc2v[4];
#pragma unroll
      for (int g = 0; g < 4; g++) {
        v8f a;
#pragma unroll
        for (int r = 0; r < 8; r++) a[r] = b2v[g];
        a = wmma16(n1k0, w2x[g][0], a);
        a = wmma16(n1k1, w2x[g][1], a);
        a = wmma16(a2,   w2h[g],    a);
        acc2v[g] = a;
      }
#pragma unroll
      for (int r = 0; r < 8; r++) {
        const float iv = sigm_f(acc2v[0][r]);
        const float fv = sigm_f(acc2v[1][r]);
        const float gv = tanh_f(acc2v[2][r]);
        const float ov = sigm_f(acc2v[3][r]);
        const float c  = fv * c2[r] + iv * gv;
        c2[r] = c;
        const float hh = ov * tanh_f(c);
        const int b    = r + 8 * g2;
        const int unit = 16 * w + nl;
        h2s[b * 32 + unit] = (_Float16)hh;
        h2f[unit * 16 + b] = hh;                  // f32 copy for projection
      }
    }
    __syncthreads();   // B3: new h2 visible

    // -------- phase D: projection y = h2 . Wp + bp (wave 3) --------
    if (w == 3 && lane < 16) {
      float y = bpv;
#pragma unroll
      for (int u = 0; u < 30; u++) y += h2f[u * 16 + lane] * wpv[u];
      out[(size_t)(batch0 + lane) * TSEQ + t] = y;
    }
    xrow += 15;
  }
}

// ---------------------------------------------------------------------------
// Launch
// ---------------------------------------------------------------------------
extern "C" void kernel_launch(void* const* d_in, const int* in_sizes, int n_in,
                              void* d_out, int out_size, void* d_ws, size_t ws_size,
                              hipStream_t stream) {
  (void)in_sizes; (void)n_in; (void)out_size; (void)ws_size;
  const float* x    = (const float*)d_in[0];
  const float* Wih1 = (const float*)d_in[1];
  const float* Whh1 = (const float*)d_in[2];
  const float* bih1 = (const float*)d_in[3];
  const float* bhh1 = (const float*)d_in[4];
  const float* Wih2 = (const float*)d_in[5];
  const float* Whh2 = (const float*)d_in[6];
  const float* bih2 = (const float*)d_in[7];
  const float* bhh2 = (const float*)d_in[8];
  const float* Wp   = (const float*)d_in[9];
  const float* bp   = (const float*)d_in[10];

  _Float16* frags = (_Float16*)d_ws;                       // 72*512 halves = 73728 B
  float*    biases = (float*)((char*)d_ws + 73728);        // 417 floats

  const int pack_threads = 72 * 512 + 512;                 // 37376
  lstm_pack_kernel<<<(pack_threads + 255) / 256, 256, 0, stream>>>(
      Wih1, Whh1, bih1, bhh1, Wih2, Whh2, bih2, bhh2, Wp, bp, frags, biases);

  lstm_fused_kernel<<<NBATCH / 16, 128, 0, stream>>>(
      x, frags, biases, (float*)d_out);
}
